// QATLinear_23553600651514
// MI455X (gfx1250) — matrix operand order repaired and checked by reading, not compiled, not run on hardware
//
#include <hip/hip_runtime.h>

typedef __attribute__((ext_vector_type(16))) __bf16        v16bf;
typedef __attribute__((ext_vector_type(8)))  float         v8f;
typedef __attribute__((ext_vector_type(4)))  unsigned int  v4u;

union FragU { v4u u[2]; v16bf bf; };

static __device__ __forceinline__ unsigned short f2bf(float f) {
    // round-to-nearest-even f32 -> bf16 (values are exactly representable anyway)
    unsigned u = __float_as_uint(f);
    u += 0x7FFFu + ((u >> 16) & 1u);
    return (unsigned short)(u >> 16);
}

// ---------------------------------------------------------------------------
// Blockwise power-of-2 int8 fake-quant + dequant, f32 -> exact bf16.
// One thread = 4 contiguous elements; 8 consecutive lanes = one 32-elem block.
// ---------------------------------------------------------------------------
__global__ __launch_bounds__(256) void qdq_to_bf16(const float* __restrict__ in,
                                                   unsigned short* __restrict__ out,
                                                   long long n4) {
    long long t = (long long)blockIdx.x * blockDim.x + threadIdx.x;
    if (t >= n4) return;
    const float4 v = ((const float4*)in)[t];

    float amax = fmaxf(fmaxf(fabsf(v.x), fabsf(v.y)), fmaxf(fabsf(v.z), fabsf(v.w)));
    // reduce amax over the 8 lanes forming one 32-element block (wave32)
    amax = fmaxf(amax, __shfl_xor(amax, 1, 32));
    amax = fmaxf(amax, __shfl_xor(amax, 2, 32));
    amax = fmaxf(amax, __shfl_xor(amax, 4, 32));

    float k = ceilf(log2f((amax + 1e-8f) / 127.0f));
    k = fminf(fmaxf(k, -128.0f), 127.0f);
    const float s  = exp2f(k);
    const float is = exp2f(-k);   // exact inverse (power of 2)

    float q0 = fminf(fmaxf(rintf(v.x * is), -128.0f), 127.0f);
    float q1 = fminf(fmaxf(rintf(v.y * is), -128.0f), 127.0f);
    float q2 = fminf(fmaxf(rintf(v.z * is), -128.0f), 127.0f);
    float q3 = fminf(fmaxf(rintf(v.w * is), -128.0f), 127.0f);

    ushort4 r;
    r.x = f2bf(q0 * s); r.y = f2bf(q1 * s); r.z = f2bf(q2 * s); r.w = f2bf(q3 * s);
    ((ushort4*)out)[t] = r;
}

// ---------------------------------------------------------------------------
// bf16 WMMA GEMM: Out[M][N] = Xh[M][K] * Wh[N][K]^T + bias[N]
// 128x128 tile per 8-wave workgroup, BK=32 (== one scale block), double-buffered
// LDS filled by GLOBAL_LOAD_ASYNC_TO_LDS_B128 (ASYNCcnt), each wave computes a
// 32x64 sub-tile with 2x4 V_WMMA_F32_16X16X32_BF16.
// ---------------------------------------------------------------------------
#define TM  128
#define TN  128
#define BK  32
#define LDT 40   // padded LDS row stride in halves (80B: 16B-aligned, bank-conflict-free)

// Copy 32 contiguous bytes (per lane) global -> LDS via the async DMA path.
// INST_OFFSET is added to BOTH the global and LDS address, so the second chunk
// is just offset:16 on the same base registers.
#define ASYNC_CP32(ldsdst, gsrc)                                            \
    asm volatile("global_load_async_to_lds_b128 %0, %1, off\n\t"            \
                 "global_load_async_to_lds_b128 %0, %1, off offset:16"      \
                 :: "v"(ldsdst), "v"(gsrc) : "memory")

#define WAIT_ASYNC0() asm volatile("s_wait_asynccnt 0x0" ::: "memory")

// LDS aperture addresses truncate to the wave-relative LDS byte offset
// (ISA 10.2: the low 32 address bits).
static __device__ __forceinline__ unsigned lds_addr32(const void* p) {
    return (unsigned)(unsigned long long)p;
}

__global__ __launch_bounds__(256) void wmma_gemm_bf16(
    const unsigned short* __restrict__ Xh,   // [M][K] bf16 bits
    const unsigned short* __restrict__ Wh,   // [N][K] bf16 bits
    const float* __restrict__ bias,          // [N]
    float* __restrict__ Out,                 // [M][N]
    int M, int N, int K) {

    __shared__ unsigned short ldsA[2][TM * LDT];
    __shared__ unsigned short ldsB[2][TN * LDT];

    const int tid  = threadIdx.x;
    const int m0   = blockIdx.y * TM;
    const int n0   = blockIdx.x * TN;

    const int wave = tid >> 5;
    const int lane = tid & 31;
    const int lr   = lane & 15;   // row/col within fragment
    const int lh   = lane >> 4;   // K-half selector
    const int wm   = wave >> 1;   // 0..3  -> 32-row band
    const int wn   = wave & 1;    // 0..1  -> 64-col band

    // global -> LDS copy mapping: thread t moves 32B (16 halves) of each tile
    const int crow   = tid >> 1;        // 0..127
    const int ccol   = (tid & 1) * 16;  // halves
    const int ldsOff = crow * LDT + ccol;

    const unsigned long long gaA =
        (unsigned long long)(Xh + (unsigned long long)(m0 + crow) * K + ccol);
    const unsigned long long gaB =
        (unsigned long long)(Wh + (unsigned long long)(n0 + crow) * K + ccol);

    const unsigned laA0 = lds_addr32(&ldsA[0][ldsOff]);
    const unsigned laA1 = lds_addr32(&ldsA[1][ldsOff]);
    const unsigned laB0 = lds_addr32(&ldsB[0][ldsOff]);
    const unsigned laB1 = lds_addr32(&ldsB[1][ldsOff]);

    v8f c[2][4];
#pragma unroll
    for (int mf = 0; mf < 2; ++mf)
#pragma unroll
        for (int nf = 0; nf < 4; ++nf) c[mf][nf] = (v8f)(0.0f);

    const int NK = K / BK;   // 128

    // prologue: DMA k-tile 0 into buffer 0
    ASYNC_CP32(laA0, gaA);
    ASYNC_CP32(laB0, gaB);
    WAIT_ASYNC0();
    __syncthreads();

    for (int kt = 0; kt < NK; ++kt) {
        const int buf = kt & 1;

        if (kt + 1 < NK) {   // DMA next k-tile into the other buffer (overlaps WMMA)
            const unsigned la = (kt & 1) ? laA0 : laA1;   // buffer (kt+1)&1
            const unsigned lb = (kt & 1) ? laB0 : laB1;
            const unsigned long long ga = gaA + (unsigned long long)(kt + 1) * (BK * 2);
            const unsigned long long gb = gaB + (unsigned long long)(kt + 1) * (BK * 2);
            ASYNC_CP32(la, ga);
            ASYNC_CP32(lb, gb);
        }

        // A fragment (16x32, MxK): lane lr = row M, holds K[lh*16 .. lh*16+15] (contiguous)
        FragU afr[2];
#pragma unroll
        for (int mf = 0; mf < 2; ++mf) {
            const int base = (wm * 32 + mf * 16 + lr) * LDT + lh * 16;
            afr[mf].u[0] = *(const v4u*)&ldsA[buf][base];
            afr[mf].u[1] = *(const v4u*)&ldsA[buf][base + 8];
        }
        // B fragment (32x16, KxN): lane lr = col N, holds K[lh*16 .. lh*16+15] (contiguous)
        FragU bfr[4];
#pragma unroll
        for (int nf = 0; nf < 4; ++nf) {
            const int base = (wn * 64 + nf * 16 + lr) * LDT + lh * 16;
            bfr[nf].u[0] = *(const v4u*)&ldsB[buf][base];
            bfr[nf].u[1] = *(const v4u*)&ldsB[buf][base + 8];
        }

#pragma unroll
        for (int mf = 0; mf < 2; ++mf)
#pragma unroll
            for (int nf = 0; nf < 4; ++nf)
                c[mf][nf] = __builtin_amdgcn_wmma_f32_16x16x32_bf16(
                    false, afr[mf].bf, false, bfr[nf].bf,
                    (short)0, c[mf][nf], false, false);

        if (kt + 1 < NK) {
            WAIT_ASYNC0();     // my DMA into the other buffer has landed
            __syncthreads();   // everyone's DMA visible; reads of `buf` complete
        }
    }

    // epilogue: C/D layout — lane lr = N, VGPR r = M = r + lh*8; add bias
#pragma unroll
    for (int nf = 0; nf < 4; ++nf) {
        const int n  = n0 + wn * 64 + nf * 16 + lr;
        const float bv = bias[n];
#pragma unroll
        for (int mf = 0; mf < 2; ++mf) {
            const int mb = m0 + wm * 32 + mf * 16 + lh * 8;
#pragma unroll
            for (int r = 0; r < 8; ++r)
                Out[(long long)(mb + r) * N + n] = c[mf][nf][r] + bv;
        }
    }
}

// ---------------------------------------------------------------------------
extern "C" void kernel_launch(void* const* d_in, const int* in_sizes, int n_in,
                              void* d_out, int out_size, void* d_ws, size_t ws_size,
                              hipStream_t stream) {
    const float* x    = (const float*)d_in[0];   // (4,2048,4096) f32
    const float* w    = (const float*)d_in[1];   // (4096,4096)   f32
    const float* bias = (const float*)d_in[2];   // (4096,)       f32

    const int N = in_sizes[2];            // 4096
    const int K = in_sizes[1] / N;        // 4096
    const int M = in_sizes[0] / K;        // 8192

    unsigned short* xh = (unsigned short*)d_ws;          // [M][K] bf16
    unsigned short* wh = xh + (size_t)M * K;             // [N][K] bf16  (total 96 MiB)

    const long long nx4 = (long long)M * K / 4;
    const long long nw4 = (long long)N * K / 4;
    qdq_to_bf16<<<(unsigned)((nx4 + 255) / 256), 256, 0, stream>>>(x, xh, nx4);
    qdq_to_bf16<<<(unsigned)((nw4 + 255) / 256), 256, 0, stream>>>(w, wh, nw4);

    dim3 grid(N / TN, M / TM);
    wmma_gemm_bf16<<<grid, 256, 0, stream>>>(xh, wh, bias, (float*)d_out, M, N, K);
}
